// SimVQ_66288525247175
// MI455X (gfx1250) — matrix operand order repaired and checked by reading, not compiled, not run on hardware
//
#include <hip/hip_runtime.h>
#include <hip/hip_bf16.h>

#define DDIM 256
#define PDIM 256
#define KCB  8192
#define MTOK 16384

#define RPB   128          // rows per block in stage 2 (8 waves x 16)
#define LDSW  264          // padded LDS row stride in halfs (528B == 4 banks shift)

typedef __attribute__((ext_vector_type(16))) __bf16 v16bf;
typedef __attribute__((ext_vector_type(8)))  float  v8f;
typedef __attribute__((ext_vector_type(4)))  unsigned int v4u;
typedef __attribute__((ext_vector_type(8)))  int  v8i;
typedef __attribute__((ext_vector_type(4)))  int  v4i;

#ifndef __has_builtin
#define __has_builtin(x) 0
#endif
#if __has_builtin(__builtin_amdgcn_tensor_load_to_lds) && \
    __has_builtin(__builtin_amdgcn_s_wait_tensorcnt)
#define HAVE_TDM 1
#else
#define HAVE_TDM 0
#endif

union BF16x16 {
    v16bf v;
    unsigned short u[16];
    uint4 q[2];
};

__device__ __forceinline__ unsigned short f2bf(float f) {
    unsigned u = __float_as_uint(f);
    unsigned r = u + 0x7FFFu + ((u >> 16) & 1u);   // round-to-nearest-even
    return (unsigned short)(r >> 16);
}

// monotonic float -> u32 map so unsigned max == float max
__device__ __forceinline__ unsigned fmap(float f) {
    unsigned u = __float_as_uint(f);
    return (u & 0x80000000u) ? ~u : (u | 0x80000000u);
}

// ---------------------------------------------------------------------------
// Stage 1: Y(bf16)[rows,256] = l2norm_rows( X[rows,256] @ W[256,256]^T )
// ---------------------------------------------------------------------------
__global__ __launch_bounds__(256) void proj_norm_kernel(
    const float* __restrict__ X, const float* __restrict__ W,
    unsigned short* __restrict__ Y)
{
    __shared__ float rowsq[16];
    const int m0    = blockIdx.x * 16;
    const int tid   = threadIdx.x;
    const int wave  = tid >> 5;
    const int lane  = tid & 31;
    const int lrow  = lane & 15;
    const int lhalf = lane >> 4;

    if (tid < 16) rowsq[tid] = 0.0f;
    __syncthreads();

    v8f acc[2];
#pragma unroll
    for (int t = 0; t < 2; ++t) {
        const int nt = wave + 8 * t;          // N-tile 0..15
        v8f c = {};
        for (int ks = 0; ks < DDIM; ks += 32) {
            BF16x16 a;
            const float* ap = X + (size_t)(m0 + lrow) * DDIM + ks + lhalf * 8;
#pragma unroll
            for (int i = 0; i < 8; ++i) a.u[i]     = f2bf(ap[i]);
#pragma unroll
            for (int i = 0; i < 8; ++i) a.u[8 + i] = f2bf(ap[16 + i]);
            BF16x16 b;
            const float* bp = W + (size_t)(nt * 16 + lrow) * DDIM + ks + lhalf * 8;
#pragma unroll
            for (int i = 0; i < 16; ++i) b.u[i] = f2bf(bp[i + (i >> 3) * 8]);
            c = __builtin_amdgcn_wmma_f32_16x16x32_bf16(
                    false, a.v, false, b.v, (short)0, c, false, false);
        }
        acc[t] = c;
    }

#pragma unroll
    for (int r = 0; r < 8; ++r)
        atomicAdd(&rowsq[r + 8 * lhalf], acc[0][r] * acc[0][r] + acc[1][r] * acc[1][r]);
    __syncthreads();

#pragma unroll
    for (int t = 0; t < 2; ++t) {
        const int col = (wave + 8 * t) * 16 + lrow;
#pragma unroll
        for (int r = 0; r < 8; ++r) {
            const int row = r + 8 * lhalf;
            const float inv = 1.0f / fmaxf(sqrtf(rowsq[row]), 1e-12f);
            Y[(size_t)(m0 + row) * PDIM + col] = f2bf(acc[t][r] * inv);
        }
    }
}

__global__ void zero_loss_kernel(float* loss) { *loss = 0.0f; }

// LDS byte offset of a __shared__ object (generic -> AS(3) -> int)
__device__ __forceinline__ unsigned lds_off(const void* p) {
    return (unsigned)(unsigned long long)
        (__attribute__((address_space(3))) const char*)p;
}

#if HAVE_TDM
// Issue one TDM DMA: 16x256 bf16 tile, global -> LDS, hardware row padding
// of 4 DWORDs every 128 DWORDs => LDS row stride 264 halfs (== LDSW).
__device__ __forceinline__ void tdm_load_tile(const unsigned short* g, unsigned lds_byte)
{
    const unsigned long long ga = (unsigned long long)(size_t)g;
    v4u g0;
    g0[0] = 1u;                                            // count=1 (user D#)
    g0[1] = lds_byte;                                      // lds_addr
    g0[2] = (unsigned)(ga & 0xFFFFFFFFull);                // global_addr[31:0]
    g0[3] = (unsigned)((ga >> 32) & 0x01FFFFFFull)         // global_addr[56:32]
          | 0x80000000u;                                   // type=2 ("image")
    v8i g1;
    g1[0] = (int)0x07910000u;   // data_size=2B | pad_enable | interval=128DW | amount=4DW
    g1[1] = (int)(256u << 16);  // tensor_dim0[15:0]=256 in [63:48]
    g1[2] = (int)(8192u << 16); // tensor_dim0 hi=0 | tensor_dim1[15:0]=8192
    g1[3] = (int)(256u << 16);  // tensor_dim1 hi=0 | tile_dim0=256
    g1[4] = 16;                 // tile_dim1=16 | tile_dim2=0
    g1[5] = 256;                // tensor_dim0_stride[31:0]=256
    g1[6] = 0;                  // stride0 hi | tensor_dim1_stride lo (unused, 2-D tile)
    g1[7] = 0;
    v4i g2 = {0, 0, 0, 0};      // tensor_dim2/3, stride2, tile_dim3: unused
    v4i g3 = {0, 0, 0, 0};
    v8i g4 = {0, 0, 0, 0, 0, 0, 0, 0};   // extra group (6-arg toolchain form)
    __builtin_amdgcn_tensor_load_to_lds(g0, g1, g2, g3, g4, 0);
}
#endif

// fallback: cooperative copy of one 16x256 bf16 tile into padded LDS
__device__ __forceinline__ void load_tile(const unsigned short* __restrict__ src,
                                          unsigned short* __restrict__ dst, int tid)
{
#pragma unroll
    for (int i = 0; i < 2; ++i) {
        const int idx = tid + i * 256;        // 512 uint4 chunks total
        const int row = idx >> 5;
        const int col = (idx & 31) * 8;
        *(uint4*)&dst[row * LDSW + col] = *(const uint4*)&src[row * 256 + col];
    }
}

// ---------------------------------------------------------------------------
// Stage 2: fused sim = fi_n @ cb_n^T -> argmax_K -> gather + loss
// block = 128 rows (8 waves x 16); cb_n tiles DMAed into LDS by the TDM
// (double-buffered), consumed by all 8 waves.
// ---------------------------------------------------------------------------
__global__ __launch_bounds__(256) void sim_argmax_kernel(
    const unsigned short* __restrict__ FiN, const unsigned short* __restrict__ CbN,
    const float* __restrict__ flat, const float* __restrict__ emb,
    float* __restrict__ outq, float* __restrict__ out_loss,
    float* __restrict__ out_idx)
{
    __shared__ __align__(16) unsigned short btile[2][16 * LDSW];   // 2 x 8.25 KB
    __shared__ unsigned long long best[RPB];
    __shared__ int   rowidx[RPB];
    __shared__ float lossred;

    const int m0    = blockIdx.x * RPB;
    const int tid   = threadIdx.x;
    const int wave  = tid >> 5;
    const int lane  = tid & 31;
    const int lrow  = lane & 15;
    const int lhalf = lane >> 4;
    const int mrow  = m0 + wave * 16;         // this wave's 16-row sub-tile

    if (tid < RPB) best[tid] = 0ull;
    if (tid == 0)  lossred = 0.0f;

    // preload the 8 A fragments for this wave's rows; register-resident
    v16bf a[8];
#pragma unroll
    for (int ks = 0; ks < 8; ++ks) {
        BF16x16 t;
        const unsigned short* ap = FiN + (size_t)(mrow + lrow) * PDIM + ks * 32 + lhalf * 8;
        t.q[0] = *(const uint4*)(ap);
        t.q[1] = *(const uint4*)(ap + 16);
        a[ks] = t.v;
    }

    float bestv[8];
    int   besti[8];
#pragma unroll
    for (int r = 0; r < 8; ++r) { bestv[r] = -3.402823466e38f; besti[r] = 0; }

#if HAVE_TDM
    const unsigned lds0 = lds_off(&btile[0][0]);
    const unsigned lds1 = lds_off(&btile[1][0]);
    if (wave == 0) {
        tdm_load_tile(CbN, lds0);
        __builtin_amdgcn_s_wait_tensorcnt(0);
    }
    __syncthreads();
#else
    load_tile(CbN, btile[0], tid);
    __syncthreads();
#endif

    for (int t = 0; t < KCB / 16; ++t) {
        const int buf = t & 1;
#if HAVE_TDM
        if (t + 1 < KCB / 16 && wave == 0)
            tdm_load_tile(CbN + (size_t)(t + 1) * 16 * PDIM, buf ? lds0 : lds1);
#else
        if (t + 1 < KCB / 16)
            load_tile(CbN + (size_t)(t + 1) * 16 * PDIM, btile[buf ^ 1], tid);
#endif

        // two independent accumulator chains (k 0..127 and 128..255)
        const unsigned short* bb = &btile[buf][lrow * LDSW + lhalf * 8];
        v8f c0 = {}, c1 = {};
#pragma unroll
        for (int ks = 0; ks < 4; ++ks) {
            BF16x16 b;
            b.q[0] = *(const uint4*)(bb + ks * 32);
            b.q[1] = *(const uint4*)(bb + ks * 32 + 16);
            c0 = __builtin_amdgcn_wmma_f32_16x16x32_bf16(
                     false, a[ks], false, b.v, (short)0, c0, false, false);
        }
#pragma unroll
        for (int ks = 4; ks < 8; ++ks) {
            BF16x16 b;
            b.q[0] = *(const uint4*)(bb + ks * 32);
            b.q[1] = *(const uint4*)(bb + ks * 32 + 16);
            c1 = __builtin_amdgcn_wmma_f32_16x16x32_bf16(
                     false, a[ks], false, b.v, (short)0, c1, false, false);
        }

        const int col = t * 16 + lrow;
#pragma unroll
        for (int r = 0; r < 8; ++r) {
            const float v = c0[r] + c1[r];
            if (v > bestv[r]) { bestv[r] = v; besti[r] = col; }
        }

#if HAVE_TDM
        if (wave == 0) __builtin_amdgcn_s_wait_tensorcnt(0);
#endif
        __syncthreads();   // one barrier/iter: publishes next tile, protects this one
    }

    // cross-lane argmax: pack (ordered value | ~idx) and u64-max in LDS.
    // ~idx makes ties resolve to the SMALLEST index, matching jnp.argmax.
#pragma unroll
    for (int r = 0; r < 8; ++r) {
        unsigned long long pk =
            ((unsigned long long)fmap(bestv[r]) << 32) |
            (unsigned long long)(0xFFFFFFFFu - (unsigned)besti[r]);
        atomicMax(&best[wave * 16 + r + 8 * lhalf], pk);
    }
    __syncthreads();

    if (tid < RPB) {
        const int bi = (int)(0xFFFFFFFFu - (unsigned)(best[tid] & 0xFFFFFFFFull));
        rowidx[tid] = bi;
        out_idx[m0 + tid] = (float)bi;
    }
    __syncthreads();

    // gather quantized rows, write q_st (== quantized forward), accumulate loss
    float lsum = 0.0f;
#pragma unroll 4
    for (int r = 0; r < RPB; ++r) {
        const float q = emb[(size_t)rowidx[r] * DDIM + tid];
        const float f = flat[(size_t)(m0 + r) * DDIM + tid];
        outq[(size_t)(m0 + r) * DDIM + tid] = q;
        const float d = q - f;
        lsum += d * d;
    }
    atomicAdd(&lossred, lsum);
    __syncthreads();
    if (tid == 0)
        atomicAdd(out_loss, lossred * (1.25f / (float)((size_t)MTOK * DDIM)));
}

// ---------------------------------------------------------------------------
extern "C" void kernel_launch(void* const* d_in, const int* in_sizes, int n_in,
                              void* d_out, int out_size, void* d_ws, size_t ws_size,
                              hipStream_t stream)
{
    const float* z    = (const float*)d_in[0];   // [B,N,D] == flat [16384,256]
    const float* emb  = (const float*)d_in[1];   // [8192,256]
    const float* W_in = (const float*)d_in[2];   // [256,256]
    const float* W_cb = (const float*)d_in[3];   // [256,256]

    unsigned short* fi_n = (unsigned short*)d_ws;                                    // 8 MB
    unsigned short* cb_n = (unsigned short*)((char*)d_ws + (size_t)MTOK * PDIM * 2); // 4 MB

    float* out      = (float*)d_out;
    float* out_q    = out;                              // 16384*256
    float* out_loss = out + (size_t)MTOK * DDIM;        // 1
    float* out_idx  = out + (size_t)MTOK * DDIM + 1;    // 16384

    proj_norm_kernel<<<MTOK / 16, 256, 0, stream>>>(z,   W_in, fi_n);
    proj_norm_kernel<<<KCB  / 16, 256, 0, stream>>>(emb, W_cb, cb_n);
    zero_loss_kernel<<<1, 1, 0, stream>>>(out_loss);
    sim_argmax_kernel<<<MTOK / RPB, 256, 0, stream>>>(fi_n, cb_n, z, emb,
                                                      out_q, out_loss, out_idx);
}
